// GNN_62251255988241
// MI455X (gfx1250) — matrix operand order repaired and checked by reading, not compiled, not run on hardware
//
#include <hip/hip_runtime.h>
#include <hip/hip_bf16.h>

// ---------------------------------------------------------------------------
// Types for CDNA5 WMMA (wave32): v_wmma_f32_16x16x32_bf16
// ---------------------------------------------------------------------------
typedef __attribute__((ext_vector_type(16))) __bf16 v16bf;
typedef __attribute__((ext_vector_type(8)))  __bf16 v8bf;
typedef __attribute__((ext_vector_type(4)))  __bf16 v4bf;
typedef __attribute__((ext_vector_type(8)))  float  v8f;

union FragBF  { v16bf v; v8bf h[2]; };
union Pack8BF { v8bf  v; v4bf h[2]; };

#define TM 128
#define TN 64
#define TK 32

__device__ __forceinline__ v4bf cvt4(float a, float b, float c, float d)
{
    v4bf r;
    r[0] = (__bf16)a; r[1] = (__bf16)b; r[2] = (__bf16)c; r[3] = (__bf16)d;
    return r;
}

// ---------------------------------------------------------------------------
// Tiled WMMA GEMM:  C[M,N] = A[M,K] @ W[K,N] (+ bias)
// fp32 inputs converted to bf16, fp32 accumulation.
// block = 256 threads (8 waves); 128x64 block tile; each wave owns a 16x64
// strip -> 4x v_wmma_f32_16x16x32_bf16 per K-step.
// LDS double-buffered; next tile's global loads are issued before the
// current tile's WMMAs so HBM/L2 latency is hidden behind matrix math.
// Edge tiles use clamped-address unconditional loads + cndmask zeroing
// (no exec-mask branching).
// ---------------------------------------------------------------------------
__global__ __launch_bounds__(256)
void wmma_gemm_kernel(const float* __restrict__ A, const float* __restrict__ W,
                      const float* __restrict__ bias, float* __restrict__ C,
                      int M, int N, int K, int lda, int ldb, int ldc)
{
    __shared__ __align__(16) __bf16 As[2][TM][TK];   // 16 KB
    __shared__ __align__(16) __bf16 Bs[2][TN][TK];   //  8 KB (transposed: [n][k])

    const int tid  = threadIdx.x;
    const int wave = tid >> 5;
    const int lane = tid & 31;
    const int l15  = lane & 15;
    const int hsel = lane >> 4;           // half-wave select

    const int mBase = blockIdx.y * TM;
    const int nBase = blockIdx.x * TN;
    const bool interiorMN = (mBase + TM <= M) && (nBase + TN <= N);

    // A-load mapping: 2 threads per row, 16 floats each
    const int arow = tid >> 1;
    const int akb  = (tid & 1) * 16;
    // B-load mapping: 8 threads per k-row, 8 floats (n) each
    const int bkk = tid >> 3;
    const int bnb = (tid & 7) * 8;

    // staged tile registers (next K-step)
    float4 a0, a1, a2, a3, b0, b1;

    auto load_stage = [&](int k0) {
        if (interiorMN && (k0 + TK <= K)) {
            // fast path: unguarded vector loads
            const float4* __restrict__ as = reinterpret_cast<const float4*>(
                A + (size_t)(mBase + arow) * lda + (k0 + akb));
            a0 = as[0]; a1 = as[1]; a2 = as[2]; a3 = as[3];
            const float4* __restrict__ bs = reinterpret_cast<const float4*>(
                W + (size_t)(k0 + bkk) * ldb + (nBase + bnb));
            b0 = bs[0]; b1 = bs[1];
        } else {
            // edge path: clamped addresses (always in-bounds), mask to zero
            const int gr = mBase + arow;
            const float* srow = A + (size_t)(gr < M ? gr : (M - 1)) * lda;
            float av[16];
            #pragma unroll
            for (int i = 0; i < 16; ++i) {
                const int kk = k0 + akb + i;
                const float x = srow[kk < K ? kk : (K - 1)];
                av[i] = (gr < M && kk < K) ? x : 0.0f;
            }
            a0 = make_float4(av[0],  av[1],  av[2],  av[3]);
            a1 = make_float4(av[4],  av[5],  av[6],  av[7]);
            a2 = make_float4(av[8],  av[9],  av[10], av[11]);
            a3 = make_float4(av[12], av[13], av[14], av[15]);

            const int gk = k0 + bkk;
            const float* wrow = W + (size_t)(gk < K ? gk : (K - 1)) * ldb;
            float bv[8];
            #pragma unroll
            for (int i = 0; i < 8; ++i) {
                const int gn = nBase + bnb + i;
                const float x = wrow[gn < N ? gn : (N - 1)];
                bv[i] = (gk < K && gn < N) ? x : 0.0f;
            }
            b0 = make_float4(bv[0], bv[1], bv[2], bv[3]);
            b1 = make_float4(bv[4], bv[5], bv[6], bv[7]);
        }
    };

    auto store_stage = [&](int buf) {
        Pack8BF p0, p1;
        p0.h[0] = cvt4(a0.x, a0.y, a0.z, a0.w);
        p0.h[1] = cvt4(a1.x, a1.y, a1.z, a1.w);
        p1.h[0] = cvt4(a2.x, a2.y, a2.z, a2.w);
        p1.h[1] = cvt4(a3.x, a3.y, a3.z, a3.w);
        *reinterpret_cast<v8bf*>(&As[buf][arow][akb])     = p0.v;
        *reinterpret_cast<v8bf*>(&As[buf][arow][akb + 8]) = p1.v;

        Bs[buf][bnb + 0][bkk] = (__bf16)b0.x;
        Bs[buf][bnb + 1][bkk] = (__bf16)b0.y;
        Bs[buf][bnb + 2][bkk] = (__bf16)b0.z;
        Bs[buf][bnb + 3][bkk] = (__bf16)b0.w;
        Bs[buf][bnb + 4][bkk] = (__bf16)b1.x;
        Bs[buf][bnb + 5][bkk] = (__bf16)b1.y;
        Bs[buf][bnb + 6][bkk] = (__bf16)b1.z;
        Bs[buf][bnb + 7][bkk] = (__bf16)b1.w;
    };

    v8f acc[4] = {};
    const int nk = (K + TK - 1) / TK;

    // prologue: stage tile 0
    load_stage(0);
    store_stage(0);
    __syncthreads();

    for (int t = 0; t < nk; ++t) {
        // issue next tile's global loads before compute (latency hiding)
        if (t + 1 < nk) load_stage((t + 1) * TK);

        const int buf = t & 1;
        // A 16x32 bf16 layout: lanes 0-15 rows M=l15, K = {0..7,16..23};
        // lanes 16-31 same rows, K = {8..15,24..31}
        FragBF fa;
        {
            const v8bf* ap = reinterpret_cast<const v8bf*>(&As[buf][wave * 16 + l15][0]);
            fa.h[0] = ap[hsel];
            fa.h[1] = ap[2 + hsel];
        }
        #pragma unroll
        for (int nt = 0; nt < 4; ++nt) {
            // B 32x16 bf16 layout: lanes 0-15 K = {0..7,16..23}, lanes 16-31 K = {8..15,24..31}
            FragBF fb;
            const v8bf* bp = reinterpret_cast<const v8bf*>(&Bs[buf][nt * 16 + l15][0]);
            fb.h[0] = bp[hsel];
            fb.h[1] = bp[2 + hsel];
            acc[nt] = __builtin_amdgcn_wmma_f32_16x16x32_bf16(
                false, fa.v, false, fb.v, (short)0, acc[nt], false, false);
        }

        if (t + 1 < nk) {
            __syncthreads();            // all waves done reading buf (t+1)&1
            store_stage((t + 1) & 1);
            __syncthreads();            // stores visible before next compute
        }
    }

    // ---- epilogue: C layout VGPR r -> M = 8*hsel + r, N = l15 ----
    const int rowBase = mBase + wave * 16 + hsel * 8;
    const bool hasb = (bias != nullptr);
    #pragma unroll
    for (int nt = 0; nt < 4; ++nt) {
        const int gc = nBase + nt * 16 + l15;
        if (gc < N) {
            const float bv = hasb ? bias[gc] : 0.0f;
            #pragma unroll
            for (int r = 0; r < 8; ++r) {
                const int gr = rowBase + r;
                if (gr < M) C[(size_t)gr * ldc + gc] = acc[nt][r] + bv;
            }
        }
    }
}

// ---------------------------------------------------------------------------
// GCN helpers
// ---------------------------------------------------------------------------
__global__ void fill_kernel(float* __restrict__ p, float v, int n)
{
    for (int i = blockIdx.x * blockDim.x + threadIdx.x; i < n;
         i += gridDim.x * blockDim.x) p[i] = v;
}

__global__ void deg_acc_kernel(const int* __restrict__ ei, const float* __restrict__ ew,
                               float* __restrict__ deg, int E)
{
    int e = blockIdx.x * blockDim.x + threadIdx.x;
    if (e < E) atomicAdd(&deg[ei[E + e]], ew[e]);   // col = ei[1][e]
}

__global__ void rsqrt_kernel(float* __restrict__ p, int n)
{
    int i = blockIdx.x * blockDim.x + threadIdx.x;
    if (i < n) p[i] = rsqrtf(p[i]);
}

// out[c,:] = xw[c,:] * dinv[c]^2 + bias  (self-loop term + bias)
__global__ void gcn_init_kernel(const float* __restrict__ xw, const float* __restrict__ dinv,
                                const float* __restrict__ bias, float* __restrict__ out,
                                int n, int F)
{
    int total = n * F;
    for (int idx = blockIdx.x * blockDim.x + threadIdx.x; idx < total;
         idx += gridDim.x * blockDim.x) {
        int i = idx / F, j = idx - i * F;
        float di = dinv[i];
        out[idx] = xw[idx] * di * di + bias[j];
    }
}

// out[col,:] += xw[row,:] * (dinv[row]*ew*dinv[col]); one block per edge
__global__ void gcn_scatter_kernel(const float* __restrict__ xw, const int* __restrict__ ei,
                                   const float* __restrict__ ew, const float* __restrict__ dinv,
                                   float* __restrict__ out, int E, int F)
{
    int e = blockIdx.x;
    int r = ei[e];
    int c = ei[E + e];
    float norm = dinv[r] * ew[e] * dinv[c];
    const float* src = xw + (size_t)r * F;
    float* dst = out + (size_t)c * F;
    for (int j = threadIdx.x * 4; j < F; j += blockDim.x * 4) {
        float4 s = *reinterpret_cast<const float4*>(src + j);
        atomicAdd(&dst[j + 0], s.x * norm);
        atomicAdd(&dst[j + 1], s.y * norm);
        atomicAdd(&dst[j + 2], s.z * norm);
        atomicAdd(&dst[j + 3], s.w * norm);
    }
}

// ---------------------------------------------------------------------------
// feature = concat(d_vecs, leaky(gcn_d[d_index]), p_emb, leaky(gcn_p[p_index]))
// ---------------------------------------------------------------------------
__global__ __launch_bounds__(256)
void assemble_feature_kernel(const int* __restrict__ d_index, const int* __restrict__ p_index,
                             const float* __restrict__ d_vecs, const float* __restrict__ p_emb,
                             const float* __restrict__ gd, const float* __restrict__ gp,
                             float* __restrict__ feat)
{
    const int i = blockIdx.x;
    const int di = d_index[i];
    const int pi = p_index[i];
    float* frow = feat + (size_t)i * 2476;
    const float* gdr = gd + (size_t)di * 1024;
    const float* gpr = gp + (size_t)pi * 128;
    for (int j = threadIdx.x; j < 300; j += 256)
        frow[j] = d_vecs[(size_t)i * 300 + j];
    for (int j = threadIdx.x; j < 1024; j += 256) {
        float x = gdr[j];
        frow[300 + j] = (x >= 0.0f) ? x : 0.01f * x;
    }
    for (int j = threadIdx.x; j < 1024; j += 256)
        frow[1324 + j] = p_emb[(size_t)i * 1024 + j];
    for (int j = threadIdx.x; j < 128; j += 256) {
        float x = gpr[j];
        frow[2348 + j] = (x >= 0.0f) ? x : 0.01f * x;
    }
}

// ---------------------------------------------------------------------------
// BatchNorm (training forward, biased variance) + activation
// ---------------------------------------------------------------------------
__global__ __launch_bounds__(256)
void bn_stats_kernel(const float* __restrict__ Z, int B, int N,
                     float* __restrict__ mean, float* __restrict__ var)
{
    __shared__ float s1[4][64];
    __shared__ float s2[4][64];
    const int c  = threadIdx.x & 63;
    const int rg = threadIdx.x >> 6;
    const int j  = blockIdx.x * 64 + c;
    float a = 0.0f, b = 0.0f;
    if (j < N) {
        for (int i = rg; i < B; i += 4) {
            float x = Z[(size_t)i * N + j];
            a += x; b += x * x;
        }
    }
    s1[rg][c] = a; s2[rg][c] = b;
    __syncthreads();
    if (rg == 0 && j < N) {
        float sa = s1[0][c] + s1[1][c] + s1[2][c] + s1[3][c];
        float sb = s2[0][c] + s2[1][c] + s2[2][c] + s2[3][c];
        float m = sa / (float)B;
        mean[j] = m;
        var[j]  = sb / (float)B - m * m;
    }
}

// act: 0 = relu, 1 = leaky(0.01)
__global__ void bn_act_kernel(float* __restrict__ Z, int B, int N,
                              const float* __restrict__ mean, const float* __restrict__ var,
                              const float* __restrict__ g, const float* __restrict__ beta,
                              int act)
{
    int total = B * N;
    for (int idx = blockIdx.x * blockDim.x + threadIdx.x; idx < total;
         idx += gridDim.x * blockDim.x) {
        int j = idx % N;
        float x = (Z[idx] - mean[j]) * rsqrtf(var[j] + 1e-5f) * g[j] + beta[j];
        x = (act == 0) ? fmaxf(x, 0.0f) : ((x >= 0.0f) ? x : 0.01f * x);
        Z[idx] = x;
    }
}

// ---------------------------------------------------------------------------
// y[i] = H[i,:] @ w (K=256) + b
// ---------------------------------------------------------------------------
__global__ __launch_bounds__(256)
void out_head_kernel(const float* __restrict__ H, const float* __restrict__ w,
                     const float* __restrict__ b, float* __restrict__ y)
{
    __shared__ float sh[256];
    const int i = blockIdx.x;
    sh[threadIdx.x] = H[(size_t)i * 256 + threadIdx.x] * w[threadIdx.x];
    __syncthreads();
    for (int s = 128; s > 0; s >>= 1) {
        if (threadIdx.x < s) sh[threadIdx.x] += sh[threadIdx.x + s];
        __syncthreads();
    }
    if (threadIdx.x == 0) y[i] = sh[0] + b[0];
}

// ---------------------------------------------------------------------------
// Launch
// ---------------------------------------------------------------------------
static inline dim3 gemm_grid(int M, int N) {
    return dim3((unsigned)((N + TN - 1) / TN), (unsigned)((M + TM - 1) / TM));
}

extern "C" void kernel_launch(void* const* d_in, const int* in_sizes, int n_in,
                              void* d_out, int out_size, void* d_ws, size_t ws_size,
                              hipStream_t stream)
{
    (void)in_sizes; (void)n_in; (void)out_size; (void)ws_size;

    const int ND = 20000, NP = 20000, ED = 200000, EP = 200000;
    const int Bn = 8192, DIM = 2476;

    const int*   d_index = (const int*)  d_in[0];
    const int*   p_index = (const int*)  d_in[1];
    const float* d_vecs  = (const float*)d_in[2];
    const float* p_emb   = (const float*)d_in[3];
    const float* d_ecfps = (const float*)d_in[4];
    const int*   d_ei    = (const int*)  d_in[5];
    const float* d_ew    = (const float*)d_in[6];
    const float* p_gos   = (const float*)d_in[7];
    const int*   p_ei    = (const int*)  d_in[8];
    const float* p_ew    = (const float*)d_in[9];
    // 10..12: hx_drug / hx_prot / h_ei -> SAGE branch is dead code (deleted in ref)
    const float* gcn_d_w = (const float*)d_in[13];
    const float* gcn_d_b = (const float*)d_in[14];
    const float* gcn_p_w = (const float*)d_in[15];
    const float* gcn_p_b = (const float*)d_in[16];
    // 17..19: sage params unused
    const float* e1_w    = (const float*)d_in[20];
    const float* e1_b    = (const float*)d_in[21];
    const float* e1_g    = (const float*)d_in[22];
    const float* e1_be   = (const float*)d_in[23];
    const float* e2_w    = (const float*)d_in[24];
    const float* e2_b    = (const float*)d_in[25];
    const float* e2_g    = (const float*)d_in[26];
    const float* e2_be   = (const float*)d_in[27];
    const float* dec1_w  = (const float*)d_in[28];
    const float* dec1_b  = (const float*)d_in[29];
    const float* dec1_g  = (const float*)d_in[30];
    const float* dec1_be = (const float*)d_in[31];
    const float* dec2_w  = (const float*)d_in[32];
    const float* dec2_b  = (const float*)d_in[33];
    const float* dec2_g  = (const float*)d_in[34];
    const float* dec2_be = (const float*)d_in[35];
    const float* o1_w    = (const float*)d_in[36];
    const float* o1_b    = (const float*)d_in[37];
    const float* o1_g    = (const float*)d_in[38];
    const float* o1_be   = (const float*)d_in[39];
    const float* o2_w    = (const float*)d_in[40];
    const float* o2_b    = (const float*)d_in[41];

    float* ws = (float*)d_ws;

    // --- workspace layout (floats), GCN phase ---
    float* XW_D = ws;                                   // 20000*1024
    float* GD   = ws + (size_t)20480000;                // 20000*1024
    float* XW_P = ws + (size_t)40960000;                // 20000*128
    float* GP   = ws + (size_t)43520000;                // 20000*128
    float* DEG_D = ws + (size_t)46080000;               // 20000 (becomes dinv)
    float* DEG_P = ws + (size_t)46100000;               // 20000 (becomes dinv)
    float* MEAN  = ws + (size_t)46120000;               // 2560
    float* VAR   = ws + (size_t)46122560;               // 2560
    // --- MLP phase (reuses GCN scratch after feature assembly) ---
    float* Z1 = ws;                                     // 8192*2048
    float* Z2 = ws + (size_t)16777216;                  // 8192*1024 (encoded)
    float* Z3 = ws + (size_t)25165824;                  // 8192*2048
    float* O1 = ws + (size_t)41943040;                  // 8192*256

    // --- output slices: (y, decoded, feature) concatenated flat ---
    float* y_out   = (float*)d_out;
    float* decoded = y_out + Bn;
    float* feat    = decoded + (size_t)Bn * DIM;

    // ================= drug GCN =================
    wmma_gemm_kernel<<<gemm_grid(ND, 1024), 256, 0, stream>>>(
        d_ecfps, gcn_d_w, nullptr, XW_D, ND, 1024, 1024, 1024, 1024, 1024);
    fill_kernel<<<(ND + 255) / 256, 256, 0, stream>>>(DEG_D, 1.0f, ND);
    deg_acc_kernel<<<(ED + 255) / 256, 256, 0, stream>>>(d_ei, d_ew, DEG_D, ED);
    rsqrt_kernel<<<(ND + 255) / 256, 256, 0, stream>>>(DEG_D, ND);
    gcn_init_kernel<<<4096, 256, 0, stream>>>(XW_D, DEG_D, gcn_d_b, GD, ND, 1024);
    gcn_scatter_kernel<<<ED, 256, 0, stream>>>(XW_D, d_ei, d_ew, DEG_D, GD, ED, 1024);

    // ================= protein GCN =================
    wmma_gemm_kernel<<<gemm_grid(NP, 128), 256, 0, stream>>>(
        p_gos, gcn_p_w, nullptr, XW_P, NP, 128, 4096, 4096, 128, 128);
    fill_kernel<<<(NP + 255) / 256, 256, 0, stream>>>(DEG_P, 1.0f, NP);
    deg_acc_kernel<<<(EP + 255) / 256, 256, 0, stream>>>(p_ei, p_ew, DEG_P, EP);
    rsqrt_kernel<<<(NP + 255) / 256, 256, 0, stream>>>(DEG_P, NP);
    gcn_init_kernel<<<1024, 256, 0, stream>>>(XW_P, DEG_P, gcn_p_b, GP, NP, 128);
    gcn_scatter_kernel<<<EP, 64, 0, stream>>>(XW_P, p_ei, p_ew, DEG_P, GP, EP, 128);

    // ================= feature concat (leaky on GCN branches) =================
    assemble_feature_kernel<<<Bn, 256, 0, stream>>>(
        d_index, p_index, d_vecs, p_emb, GD, GP, feat);

    // ================= encoder =================
    wmma_gemm_kernel<<<gemm_grid(Bn, 2048), 256, 0, stream>>>(
        feat, e1_w, e1_b, Z1, Bn, 2048, DIM, DIM, 2048, 2048);
    bn_stats_kernel<<<(2048 + 63) / 64, 256, 0, stream>>>(Z1, Bn, 2048, MEAN, VAR);
    bn_act_kernel<<<2048, 256, 0, stream>>>(Z1, Bn, 2048, MEAN, VAR, e1_g, e1_be, 0);

    wmma_gemm_kernel<<<gemm_grid(Bn, 1024), 256, 0, stream>>>(
        Z1, e2_w, e2_b, Z2, Bn, 1024, 2048, 2048, 1024, 1024);
    bn_stats_kernel<<<(1024 + 63) / 64, 256, 0, stream>>>(Z2, Bn, 1024, MEAN, VAR);
    bn_act_kernel<<<1024, 256, 0, stream>>>(Z2, Bn, 1024, MEAN, VAR, e2_g, e2_be, 0);

    // ================= decoder =================
    wmma_gemm_kernel<<<gemm_grid(Bn, 2048), 256, 0, stream>>>(
        Z2, dec1_w, dec1_b, Z3, Bn, 2048, 1024, 1024, 2048, 2048);
    bn_stats_kernel<<<(2048 + 63) / 64, 256, 0, stream>>>(Z3, Bn, 2048, MEAN, VAR);
    bn_act_kernel<<<2048, 256, 0, stream>>>(Z3, Bn, 2048, MEAN, VAR, dec1_g, dec1_be, 0);

    wmma_gemm_kernel<<<gemm_grid(Bn, DIM), 256, 0, stream>>>(
        Z3, dec2_w, dec2_b, decoded, Bn, DIM, 2048, 2048, DIM, DIM);
    bn_stats_kernel<<<(DIM + 63) / 64, 256, 0, stream>>>(decoded, Bn, DIM, MEAN, VAR);
    bn_act_kernel<<<2048, 256, 0, stream>>>(decoded, Bn, DIM, MEAN, VAR, dec2_g, dec2_be, 0);

    // ================= prediction head =================
    wmma_gemm_kernel<<<gemm_grid(Bn, 256), 256, 0, stream>>>(
        Z2, o1_w, o1_b, O1, Bn, 256, 1024, 1024, 256, 256);
    bn_stats_kernel<<<(256 + 63) / 64, 256, 0, stream>>>(O1, Bn, 256, MEAN, VAR);
    bn_act_kernel<<<512, 256, 0, stream>>>(O1, Bn, 256, MEAN, VAR, o1_g, o1_be, 1);

    out_head_kernel<<<Bn, 256, 0, stream>>>(O1, o2_w, o2_b, y_out);
}
